// NTXentLoss_38457137168752
// MI455X (gfx1250) — matrix operand order repaired and checked by reading, not compiled, not run on hardware
//
#include <hip/hip_runtime.h>
#include <hip/hip_bf16.h>

typedef __attribute__((ext_vector_type(16))) _Float16 v16h;
typedef __attribute__((ext_vector_type(8)))  _Float16 v8h;
typedef __attribute__((ext_vector_type(4)))  _Float16 v4h;
typedef __attribute__((ext_vector_type(8)))  float    v8f;

#define BATCH   4096
#define NROWS   8192      // 2*BATCH
#define DDIM    128
#define NEG_BIG (-1e9f)

// zn is pre-scaled by QSCALE = sqrt(2 * log2(e)), so the WMMA accumulator holds
//   acc = log2(e) * (2 * cosine)  ==  log2(e) * logits
// and exp(logits) == exp2(acc): the inner loop needs ONLY v_exp_f32 + add.
#define QSCALE  1.69864364f          // sqrt(2 * 1.4426950408889634)
#define LN2     0.69314718055994531f // recover natural-log logits at the end

#if __has_builtin(__builtin_amdgcn_exp2f)
#define FAST_EXP2(x) __builtin_amdgcn_exp2f(x)
#else
#define FAST_EXP2(x) exp2f(x)
#endif

// ---------------------------------------------------------------------------
// Kernel 1: fused concat + L2-normalize + temperature/log2e pre-scale + f16.
// One wave per row, lane handles 4 contiguous floats.
// ---------------------------------------------------------------------------
__global__ void __launch_bounds__(256)
ntxent_normalize(const float* __restrict__ zi, const float* __restrict__ zj,
                 _Float16* __restrict__ zn) {
    const int wave = (blockIdx.x * blockDim.x + threadIdx.x) >> 5;   // row id
    const int lane = threadIdx.x & 31;

    const float* src = (wave < BATCH) ? (zi + (size_t)wave * DDIM)
                                      : (zj + (size_t)(wave - BATCH) * DDIM);
    const float4 v = ((const float4*)src)[lane];
    float ss = v.x * v.x + v.y * v.y + v.z * v.z + v.w * v.w;
    #pragma unroll
    for (int m = 16; m >= 1; m >>= 1) ss += __shfl_xor(ss, m, 32);
    const float inv = QSCALE / fmaxf(sqrtf(ss), 1e-8f);  // cosine_similarity eps

    v4h o;
    o[0] = (_Float16)(v.x * inv);
    o[1] = (_Float16)(v.y * inv);
    o[2] = (_Float16)(v.z * inv);
    o[3] = (_Float16)(v.w * inv);
    ((v4h*)(zn + (size_t)wave * DDIM))[lane] = o;
}

// ---------------------------------------------------------------------------
// Kernel 2: fused GEMM (zn @ zn^T) + streaming softmax denominator + positive
// gather. One wave per 16-row tile, A in registers, 2 column tiles (16x32 of
// output) per iteration -> 8 independent-chain WMMAs per loop to overlap with
// the B-tile global loads. Diagonal mask / positive gather only happen in two
// tile-aligned windows -> hoisted to a rare slow path (uniform branch).
//
// WMMA operand layouts used here (16-bit, wave32; A/B k order is contiguous, same for both):
//   A (16x32):  lane = M (mod 16); half0 K=kb+0..15, half1 K=kb+16..31
//   B (32x16):  lane = N (mod 16); half0 K=kb+0..15, half1 K=kb+16..31
//   C/D (16x16 f32): VGPR r, lanes 0-15 -> (M=r, N=lane),
//                            lanes 16-31 -> (M=r+8, N=lane-16)
// ---------------------------------------------------------------------------
__global__ void __launch_bounds__(256)
ntxent_fused(const _Float16* __restrict__ zn, float* __restrict__ partials) {
    const int lane  = threadIdx.x & 31;
    const int wave  = blockIdx.x * (blockDim.x >> 5) + (threadIdx.x >> 5);
    const int rbase = wave * 16;                 // first row of this wave's tile
    const int half  = lane >> 4;
    const int l15   = lane & 15;

    // ---- A operand: 16 rows x 128 K resident in 4 x v16h (32 VGPRs) ----
    const _Float16* arow = zn + (size_t)(rbase + l15) * DDIM;
    v16h A[4];
    #pragma unroll
    for (int kc = 0; kc < 4; ++kc) {
        const int kb = kc * 32;
        union { v16h v; v8h h[2]; } u;
        u.h[0] = *(const v8h*)(arow + kb + half * 16);
        u.h[1] = *(const v8h*)(arow + kb + half * 16 + 8);
        A[kc] = u.v;
    }

    // Wave-uniform (scalarized) windows that need per-element handling:
    const int rb_u     = __builtin_amdgcn_readfirstlane(rbase);
    const int diag_win = rb_u & ~31;                       // window with diagonal
    const int pos_win  = (rb_u >= BATCH) ? ((rb_u - BATCH) & ~31) : -1;

    float s[8], pos[8];
    #pragma unroll
    for (int r = 0; r < 8; ++r) { s[r] = 0.0f; pos[r] = 0.0f; }

    for (int cb = 0; cb < NROWS; cb += 32) {
        const _Float16* b0 = zn + (size_t)(cb + l15) * DDIM + half * 16;
        const _Float16* b1 = b0 + 16 * DDIM;
        if (cb + 32 < NROWS) {
            __builtin_prefetch(b0 + 32 * DDIM, 0, 1);      // global_prefetch_b8
            __builtin_prefetch(b1 + 32 * DDIM, 0, 1);
        }

        v8f acc0 = {}, acc1 = {};
        #pragma unroll
        for (int kc = 0; kc < 4; ++kc) {
            const v16h B0 = *(const v16h*)(b0 + kc * 32);
            const v16h B1 = *(const v16h*)(b1 + kc * 32);
            acc0 = __builtin_amdgcn_wmma_f32_16x16x32_f16(
                       false, A[kc], false, B0, (short)0, acc0, false, false);
            acc1 = __builtin_amdgcn_wmma_f32_16x16x32_f16(
                       false, A[kc], false, B1, (short)0, acc1, false, false);
        }

        if (__builtin_expect(cb != diag_win && cb != pos_win, 1)) {
            // ---- hot path: 254/256 iterations, pure exp2 + accumulate ----
            #pragma unroll
            for (int r = 0; r < 8; ++r) {
                s[r] += FAST_EXP2(acc0[r]);
                s[r] += FAST_EXP2(acc1[r]);
            }
        } else {
            // ---- rare path: diagonal mask and/or positive-logit gather ----
            #pragma unroll
            for (int t = 0; t < 2; ++t) {
                const v8f acc  = t ? acc1 : acc0;
                const int gcol = cb + t * 16 + l15;
                #pragma unroll
                for (int r = 0; r < 8; ++r) {
                    const int grow = rbase + r + 8 * half;
                    float e = FAST_EXP2(acc[r]);
                    if (gcol == grow) e = 0.0f;            // diag -> -1e9 -> exp 0
                    s[r] += e;
                    if (grow >= BATCH && gcol == grow - BATCH) pos[r] = acc[r];
                }
            }
        }
    }

    // Row-wise reduce across the 16 lanes of each half (rows are disjoint
    // between halves, so width-16 xor shuffles keep them separate).
    #pragma unroll
    for (int r = 0; r < 8; ++r) {
        #pragma unroll
        for (int m = 8; m >= 1; m >>= 1) {
            s[r]   += __shfl_xor(s[r],   m, 16);
            pos[r] += __shfl_xor(pos[r], m, 16);
        }
    }

    // loss_row = ln(sum exp) - logit[row, label[row]]   (pos is in exp2 domain)
    float local = 0.0f;
    if (l15 == 0) {
        #pragma unroll
        for (int r = 0; r < 8; ++r) {
            const int grow = rbase + r + 8 * half;
            const float p  = (grow < BATCH) ? NEG_BIG : pos[r] * LN2;
            local += __logf(s[r]) - p;
        }
    }
    const float other = __shfl(local, 16, 32);  // fold lane16 (rows 8-15) into lane0
    if (lane == 0) partials[wave] = local + other;
}

// ---------------------------------------------------------------------------
// Kernel 3: deterministic final reduction of 512 per-tile partials -> mean.
// ---------------------------------------------------------------------------
__global__ void __launch_bounds__(256)
ntxent_reduce(const float* __restrict__ partials, float* __restrict__ out) {
    __shared__ float sm[256];
    const int t = threadIdx.x;
    sm[t] = partials[t] + partials[t + 256];
    __syncthreads();
    #pragma unroll
    for (int st = 128; st >= 1; st >>= 1) {
        if (t < st) sm[t] += sm[t + st];
        __syncthreads();
    }
    if (t == 0) out[0] = sm[0] * (1.0f / (float)NROWS);
}

// ---------------------------------------------------------------------------
extern "C" void kernel_launch(void* const* d_in, const int* in_sizes, int n_in,
                              void* d_out, int out_size, void* d_ws, size_t ws_size,
                              hipStream_t stream) {
    const float* zi = (const float*)d_in[0];
    const float* zj = (const float*)d_in[1];
    float* out      = (float*)d_out;

    // workspace layout: [0, 2MB) f16 scaled-normalized embeddings; 512 partials
    _Float16* zn    = (_Float16*)d_ws;
    float* partials = (float*)((char*)d_ws + (size_t)NROWS * DDIM * sizeof(_Float16));

    // 8192 rows, 1 wave/row, 8 waves/block -> 1024 blocks
    ntxent_normalize<<<NROWS / 8, 256, 0, stream>>>(zi, zj, zn);

    // 512 row-tiles, 1 wave/tile, 8 waves/block -> 64 blocks
    ntxent_fused<<<(NROWS / 16) / 8, 256, 0, stream>>>(zn, partials);

    // 512 partials -> scalar loss
    ntxent_reduce<<<1, 256, 0, stream>>>(partials, out);
}
